// Devign2Linear_33243046871385
// MI455X (gfx1250) — matrix operand order repaired and checked by reading, not compiled, not run on hardware
//
#include <hip/hip_runtime.h>
#include <hip/hip_bf16.h>

// ---------------------------------------------------------------------------
// Devign GGNN forward, CDNA5 (gfx1250) WMMA implementation.
//  - all heavy GEMMs through v_wmma_f32_16x16x32_bf16 (wave32, 16x16x32)
//  - B weight tiles staged via GLOBAL_LOAD_ASYNC_TO_LDS_B128 (ASYNCcnt)
//  - A activation stream prefetched with global_prefetch_b8
// ---------------------------------------------------------------------------

#define N_NODES  100000
#define N_EDGES  320000
#define N_GRAPHS 200
#define EMB      128
#define DD       256
#define STEPS    6
#define HID      128
#define D3       768   // 3*DD

typedef __attribute__((ext_vector_type(16))) __bf16 v16bf;
typedef __attribute__((ext_vector_type(8)))  float  v8f;

union V16 { v16bf v; uint4 q[2]; };

static __device__ __forceinline__ unsigned short f32_to_bf16(float f) {
    unsigned int u = __float_as_uint(f);
    unsigned int r = u + 0x7FFFu + ((u >> 16) & 1u);   // round-to-nearest-even
    return (unsigned short)(r >> 16);
}

static __device__ __forceinline__ float sigmoidf_dev(float x) {
    return 1.0f / (1.0f + __expf(-x));
}

// Async copy: 16 bytes global -> LDS, tracked by ASYNCcnt (CDNA5 §10.7/§15.18).
//   lds_off : byte offset within workgroup LDS allocation (VGPR)
//   voff    : per-lane 32-bit byte offset added to saddr (GVS mode)
//   base    : uniform 64-bit global base address (SGPR pair)
static __device__ __forceinline__ void async_ld_b128(unsigned lds_off,
                                                     unsigned voff,
                                                     unsigned long long base) {
    asm volatile("global_load_async_to_lds_b128 %0, %1, %2"
                 :: "v"(lds_off), "v"(voff), "s"(base) : "memory");
}
static __device__ __forceinline__ void wait_async0() {
    asm volatile("s_wait_asynccnt 0x0" ::: "memory");
}

// ---------------------------------------------------------------------------
// Weight preparation: fp32 -> bf16 column-major Bt[n][k]
// ---------------------------------------------------------------------------
__global__ void dv_transpose_bf16(const float* __restrict__ src,
                                  unsigned short* __restrict__ dst,
                                  int R, int C) {
    // src is [R][C]; dst is [C][R]:  dst[c*R + r] = bf16(src[r*C + c])
    int gid = blockIdx.x * blockDim.x + threadIdx.x;
    if (gid >= R * C) return;
    int r = gid / C, c = gid % C;
    dst[(size_t)c * R + r] = f32_to_bf16(src[gid]);
}

__global__ void dv_copy_bf16(const float* __restrict__ src,
                             unsigned short* __restrict__ dst, int n) {
    int gid = blockIdx.x * blockDim.x + threadIdx.x;
    if (gid < n) dst[gid] = f32_to_bf16(src[gid]);
}

__global__ void dv_zero_f32(float* __restrict__ p, int n) {
    int gid = blockIdx.x * blockDim.x + threadIdx.x;
    if (gid < n) p[gid] = 0.0f;
}

// ---------------------------------------------------------------------------
// WMMA GEMM:  C[nrows, ncols] = A[nrows, K](fp32) * Bt(bf16, col-major [n][k])
//             + bias.  Block: 256 threads (8 wave32), tile 64(M) x 128(N),
// K-chunk 32.  Each wave: 32x32 region = 2x2 wmma accumulators.
// B tile staged with async global->LDS B128 copies; A converted in-register.
// ---------------------------------------------------------------------------
#define LDA 40   // padded LDS row stride (bf16): 80B = 20 dwords, 16B-aligned
__global__ __launch_bounds__(256)
void dv_gemm_bf16(const float* __restrict__ A,
                  const unsigned short* __restrict__ Bt,  // [ncols][K]
                  const float* __restrict__ bias,
                  float* __restrict__ C,
                  int nrows, int K, int ncols) {
    __shared__ unsigned short As[64 * LDA];    // [row][k] tile, fp32->bf16
    __shared__ unsigned short Bs[128 * LDA];   // [col][k] tile

    const int t    = threadIdx.x;
    const int lane = t & 31;
    const int wid  = t >> 5;
    const int wM   = (wid & 1) * 32;           // wave row offset in block tile
    const int wN   = (wid >> 1) * 32;          // wave col offset in block tile
    const int lr   = lane & 15;
    const int hi   = lane >> 4;                // lane half (ISA fragment layout)
    const int rowBlk = blockIdx.x * 64;
    const int colBlk = blockIdx.y * 128;

    v8f acc[2][2] = {};

    // A staging assignment: thread -> (row, 8 consecutive k)
    const int aRow = t >> 2;
    const int aK   = (t & 3) * 8;
    const int gRow = rowBlk + aRow;

    const unsigned long long bAddr = (unsigned long long)(uintptr_t)Bt;
    const unsigned bsBase = (unsigned)(uintptr_t)&Bs[0];   // LDS byte offset

    for (int k0 = 0; k0 < K; k0 += 32) {
        // ---- issue async B-tile copies: 128 cols x 32 k = 512 x 16B -------
        {
            const unsigned c0 = (unsigned)t * 2u;
            #pragma unroll
            for (int j = 0; j < 2; ++j) {
                unsigned c  = c0 + j;
                unsigned nn = c >> 2;                  // 0..127
                unsigned ch = c & 3;                   // 16B chunk within col
                unsigned voff =
                    (unsigned)(((colBlk + nn) * K + k0) * 2 + ch * 16);
                unsigned loff = bsBase + nn * (LDA * 2) + ch * 16;
                async_ld_b128(loff, voff, bAddr);
            }
        }
        // ---- stage A: 64x32 fp32 -> bf16 into LDS -------------------------
        {
            float tmp[8];
            if (gRow < nrows) {
                const float4* p =
                    (const float4*)(A + (size_t)gRow * K + k0 + aK);
                float4 f0 = p[0], f1 = p[1];
                tmp[0] = f0.x; tmp[1] = f0.y; tmp[2] = f0.z; tmp[3] = f0.w;
                tmp[4] = f1.x; tmp[5] = f1.y; tmp[6] = f1.z; tmp[7] = f1.w;
                if (k0 + 32 < K)   // prefetch next A chunk (global_prefetch_b8)
                    __builtin_prefetch(A + (size_t)gRow * K + k0 + 32 + aK, 0, 1);
            } else {
                #pragma unroll
                for (int j = 0; j < 8; ++j) tmp[j] = 0.0f;
            }
            #pragma unroll
            for (int j = 0; j < 8; ++j)
                As[aRow * LDA + aK + j] = f32_to_bf16(tmp[j]);
        }
        wait_async0();        // this wave's async LDS writes have landed
        __syncthreads();      // all waves' ds stores + asyncs visible

        // ---- build fragments per ISA VGPR layout --------------------------
        V16 a[2], b[2];
        #pragma unroll
        for (int sm = 0; sm < 2; ++sm) {
            const unsigned short* p = &As[(wM + sm * 16 + lr) * LDA];
            // elements 0..7 : k = hi*8 + e ; elements 8..15 : k = 16+hi*8+e
            a[sm].q[0] = *(const uint4*)(p + hi * 8);
            a[sm].q[1] = *(const uint4*)(p + 16 + hi * 8);
        }
        #pragma unroll
        for (int sn = 0; sn < 2; ++sn) {
            const unsigned short* p = &Bs[(wN + sn * 16 + lr) * LDA];
            // elements 0..7 : k = hi*8 + e ; elements 8..15 : k = 16+hi*8+e
            b[sn].q[0] = *(const uint4*)(p + hi * 8);
            b[sn].q[1] = *(const uint4*)(p + 16 + hi * 8);
        }
        #pragma unroll
        for (int sm = 0; sm < 2; ++sm)
            #pragma unroll
            for (int sn = 0; sn < 2; ++sn)
                acc[sm][sn] = __builtin_amdgcn_wmma_f32_16x16x32_bf16(
                    false, a[sm].v, false, b[sn].v,
                    (short)0, acc[sm][sn], false, false);
        __syncthreads();
    }

    // ---- store: C/D layout -> row = v + hi*8, col = lr --------------------
    #pragma unroll
    for (int sm = 0; sm < 2; ++sm) {
        #pragma unroll
        for (int sn = 0; sn < 2; ++sn) {
            int col  = colBlk + wN + sn * 16 + lr;
            float bv = bias ? bias[col] : 0.0f;
            #pragma unroll
            for (int v = 0; v < 8; ++v) {
                int row = rowBlk + wM + sm * 16 + hi * 8 + v;
                if (row < nrows)
                    C[(size_t)row * ncols + col] = acc[sm][sn][v] + bv;
            }
        }
    }
}

// ---------------------------------------------------------------------------
// Edge scatter-add: agg[dst] += m[src], D=256 floats per node.
// ---------------------------------------------------------------------------
__global__ void dv_scatter_add(const float* __restrict__ m,
                               const long long* __restrict__ src,
                               const long long* __restrict__ dst,
                               float* __restrict__ agg) {
    int gid = blockIdx.x * blockDim.x + threadIdx.x;
    int e   = gid >> 6;
    if (e >= N_EDGES) return;
    int c4 = (gid & 63) << 2;
    long long s = src[e], d = dst[e];
    float4 v = *(const float4*)(m + (size_t)s * DD + c4);
    float* o = agg + (size_t)d * DD + c4;
    atomicAdd(o + 0, v.x);
    atomicAdd(o + 1, v.y);
    atomicAdd(o + 2, v.z);
    atomicAdd(o + 3, v.w);
}

// ---------------------------------------------------------------------------
// GRU cell elementwise:  h = (1-z)*n + z*h  (gi/gh already contain biases)
// ---------------------------------------------------------------------------
__global__ void dv_gru_elem(const float* __restrict__ gi,
                            const float* __restrict__ gh,
                            float* __restrict__ h) {
    int gid = blockIdx.x * blockDim.x + threadIdx.x;
    if (gid >= N_NODES * DD) return;
    int i = gid / DD, c = gid % DD;
    size_t b3 = (size_t)i * D3;
    float ir = gi[b3 + c], iz = gi[b3 + DD + c], in_ = gi[b3 + 2 * DD + c];
    float hr = gh[b3 + c], hz = gh[b3 + DD + c], hn  = gh[b3 + 2 * DD + c];
    float r = sigmoidf_dev(ir + hr);
    float z = sigmoidf_dev(iz + hz);
    float n = tanhf(in_ + r * hn);
    size_t idx = (size_t)i * DD + c;
    h[idx] = (1.0f - z) * n + z * h[idx];
}

// ---------------------------------------------------------------------------
// Global mean pool (sums + counts via atomics)
// ---------------------------------------------------------------------------
__global__ void dv_pool_sum(const float* __restrict__ h,
                            const long long* __restrict__ batch,
                            float* __restrict__ sums) {
    int gid = blockIdx.x * blockDim.x + threadIdx.x;
    int i   = gid >> 6;
    if (i >= N_NODES) return;
    int c4 = (gid & 63) << 2;
    long long g = batch[i];
    float4 v = *(const float4*)(h + (size_t)i * DD + c4);
    float* o = sums + (size_t)g * DD + c4;
    atomicAdd(o + 0, v.x);
    atomicAdd(o + 1, v.y);
    atomicAdd(o + 2, v.z);
    atomicAdd(o + 3, v.w);
}

__global__ void dv_pool_count(const long long* __restrict__ batch,
                              float* __restrict__ cnts) {
    int i = blockIdx.x * blockDim.x + threadIdx.x;
    if (i < N_NODES) atomicAdd(&cnts[batch[i]], 1.0f);
}

// ---------------------------------------------------------------------------
// Classifier head: relu(repr@W1^T+b1) @ W2^T + b2 -> sigmoid.  Tiny (200x256).
// ---------------------------------------------------------------------------
__global__ __launch_bounds__(128)
void dv_classifier(const float* __restrict__ sums,
                   const float* __restrict__ cnts,
                   const float* __restrict__ w1, const float* __restrict__ b1,
                   const float* __restrict__ w2, const float* __restrict__ b2,
                   float* __restrict__ out) {
    __shared__ float repr[DD];
    __shared__ float red[HID];
    int g = blockIdx.x, t = threadIdx.x;
    float inv = 1.0f / fmaxf(cnts[g], 1.0f);
    repr[t]       = sums[(size_t)g * DD + t] * inv;
    repr[t + 128] = sums[(size_t)g * DD + 128 + t] * inv;
    __syncthreads();
    float acc = b1[t];
    #pragma unroll 4
    for (int k = 0; k < DD; ++k) acc += repr[k] * w1[(size_t)t * DD + k];
    red[t] = fmaxf(acc, 0.0f) * w2[t];
    __syncthreads();
    for (int s = HID / 2; s > 0; s >>= 1) {
        if (t < s) red[t] += red[t + s];
        __syncthreads();
    }
    if (t == 0) out[g] = sigmoidf_dev(red[0] + b2[0]);
}

// ---------------------------------------------------------------------------
// Host-side orchestration
// ---------------------------------------------------------------------------
static void launch_gemm(const float* A, const unsigned short* Bt,
                        const float* bias, float* C,
                        int nrows, int K, int ncols, hipStream_t s) {
    dim3 grid((nrows + 63) / 64, ncols / 128);
    dv_gemm_bf16<<<grid, dim3(256), 0, s>>>(A, Bt, bias, C, nrows, K, ncols);
}

extern "C" void kernel_launch(void* const* d_in, const int* in_sizes, int n_in,
                              void* d_out, int out_size, void* d_ws, size_t ws_size,
                              hipStream_t stream) {
    const float* x      = (const float*)d_in[0];
    const float* lin_w  = (const float*)d_in[1];   // [256,128] = [n][k]
    const float* lin_b  = (const float*)d_in[2];   // [256]
    const float* ggnn_w = (const float*)d_in[3];   // [6,256,256] = [k][n]
    const float* w_ih   = (const float*)d_in[4];   // [768,256] = [n][k]
    const float* w_hh   = (const float*)d_in[5];   // [768,256] = [n][k]
    const float* b_ih   = (const float*)d_in[6];   // [768]
    const float* b_hh   = (const float*)d_in[7];   // [768]
    const float* cw1    = (const float*)d_in[8];
    const float* cb1    = (const float*)d_in[9];
    const float* cw2    = (const float*)d_in[10];
    const float* cb2    = (const float*)d_in[11];
    const long long* ei = (const long long*)d_in[12]; // [2, E]
    const long long* bat= (const long long*)d_in[13]; // [N]
    float* out = (float*)d_out;

    const long long* e_src = ei;            // edge_index[0]
    const long long* e_dst = ei + N_EDGES;  // edge_index[1]

    // -------- workspace carve-out (all bf16 weights col-major [n][k]) ------
    char* w = (char*)d_ws;
    unsigned short* linBt  = (unsigned short*)w; w += (size_t)DD * EMB * 2;
    unsigned short* ggnnBt = (unsigned short*)w; w += (size_t)STEPS * DD * DD * 2;
    unsigned short* wihBt  = (unsigned short*)w; w += (size_t)D3 * DD * 2;
    unsigned short* whhBt  = (unsigned short*)w; w += (size_t)D3 * DD * 2;
    float* h    = (float*)w; w += (size_t)N_NODES * DD * 4;
    float* m    = (float*)w; w += (size_t)N_NODES * DD * 4;
    float* agg  = (float*)w; w += (size_t)N_NODES * DD * 4;
    float* gi   = (float*)w; w += (size_t)N_NODES * D3 * 4;
    float* gh   = (float*)w; w += (size_t)N_NODES * D3 * 4;
    float* sums = (float*)w; w += (size_t)N_GRAPHS * DD * 4;
    float* cnts = (float*)w; w += (size_t)N_GRAPHS * 4;

    // -------- weight conversion (tiny, once per launch) --------------------
    {
        int n = DD * EMB;   // lin_w already [n][k] -> straight convert
        dv_copy_bf16<<<(n + 255) / 256, 256, 0, stream>>>(lin_w, linBt, n);
        for (int s = 0; s < STEPS; ++s) {   // ggnn_w[s] is [k][n] -> transpose
            dv_transpose_bf16<<<(DD * DD + 255) / 256, 256, 0, stream>>>(
                ggnn_w + (size_t)s * DD * DD, ggnnBt + (size_t)s * DD * DD,
                DD, DD);
        }
        n = D3 * DD;        // gru weights already [n][k] -> straight convert
        dv_copy_bf16<<<(n + 255) / 256, 256, 0, stream>>>(w_ih, wihBt, n);
        dv_copy_bf16<<<(n + 255) / 256, 256, 0, stream>>>(w_hh, whhBt, n);
    }

    // -------- h = x @ lin_w^T + lin_b --------
    launch_gemm(x, linBt, lin_b, h, N_NODES, EMB, DD, stream);

    // -------- GGNN steps --------
    const int nAgg  = N_NODES * DD;
    const int nScat = N_EDGES * 64;
    for (int s = 0; s < STEPS; ++s) {
        launch_gemm(h, ggnnBt + (size_t)s * DD * DD, nullptr, m,
                    N_NODES, DD, DD, stream);
        dv_zero_f32<<<(nAgg + 255) / 256, 256, 0, stream>>>(agg, nAgg);
        dv_scatter_add<<<(nScat + 255) / 256, 256, 0, stream>>>(m, e_src, e_dst, agg);
        launch_gemm(agg, wihBt, b_ih, gi, N_NODES, DD, D3, stream);
        launch_gemm(h,   whhBt, b_hh, gh, N_NODES, DD, D3, stream);
        dv_gru_elem<<<(nAgg + 255) / 256, 256, 0, stream>>>(gi, gh, h);
    }

    // -------- mean pool + classifier --------
    {
        int n = N_GRAPHS * DD;
        dv_zero_f32<<<(n + 255) / 256, 256, 0, stream>>>(sums, n);
        dv_zero_f32<<<1, 256, 0, stream>>>(cnts, N_GRAPHS);
        int nPool = N_NODES * 64;
        dv_pool_sum<<<(nPool + 255) / 256, 256, 0, stream>>>(h, bat, sums);
        dv_pool_count<<<(N_NODES + 255) / 256, 256, 0, stream>>>(bat, cnts);
        dv_classifier<<<N_GRAPHS, HID, 0, stream>>>(sums, cnts, cw1, cb1, cw2, cb2, out);
    }
}
